// PointNet2ClassificationSSG_WOPL_45973329936990
// MI455X (gfx1250) — compile-verified, matrix operand order checked by reading, not run on hardware
//
#include <hip/hip_runtime.h>
#include <hip/hip_bf16.h>
#include <math.h>

// ---------------------------------------------------------------------------
// Types for CDNA5 WMMA (wave32): 16x16x32 f16 -> f32
// ---------------------------------------------------------------------------
typedef __attribute__((ext_vector_type(16))) _Float16 v16h;
typedef __attribute__((ext_vector_type(8)))  _Float16 v8h;
typedef __attribute__((ext_vector_type(8)))  float    v8f;

__device__ __forceinline__ v8f wmma_f16(v16h a, v16h b, v8f c) {
    return __builtin_amdgcn_wmma_f32_16x16x32_f16(
        /*neg_a=*/false, a, /*neg_b=*/false, b,
        /*c_mod=*/(short)0, c, /*reuse_a=*/false, /*reuse_b=*/false);
}

// Load one 16(x)K=32 fragment from a row-major f16 matrix (ld elements/row).
// A-operand: lane<16 holds row (lane), K {kb..kb+7, kb+16..kb+23};
//            lane>=16 holds row (lane-16), K {kb+8..kb+15, kb+24..kb+31}.
// Weight matrix W[n][K] row-major serves as the B operand with identical
// addressing (lane indexes the N column, contiguous K per lane).
__device__ __forceinline__ v16h load_frag(const _Float16* base, int ld) {
    int lane = threadIdx.x & 31;
    int half = lane >> 4;
    int r    = lane & 15;
    const _Float16* p = base + (size_t)r * ld + half * 8;
    v8h lo = *(const v8h*)(p);
    v8h hi = *(const v8h*)(p + 16);
    return __builtin_shufflevector(lo, hi, 0,1,2,3,4,5,6,7,8,9,10,11,12,13,14,15);
}

// ---------------------------------------------------------------------------
// Weight prep: f32 W[co][ci] -> f16 Wp[co][kpad] zero-padded, plus BN scale/bias
// ---------------------------------------------------------------------------
__global__ void prep_layer_kernel(const float* __restrict__ W,
                                  const float* __restrict__ g,
                                  const float* __restrict__ b,
                                  _Float16* __restrict__ Wp,
                                  float* __restrict__ scl,
                                  float* __restrict__ bias,
                                  int co, int ci, int kpad) {
    int t = blockIdx.x * blockDim.x + threadIdx.x;
    if (t >= co * kpad) return;
    int o = t / kpad, k = t - o * kpad;
    Wp[t] = (k < ci) ? (_Float16)W[(size_t)o * ci + k] : (_Float16)0.0f;
    if (k == 0) {
        scl[o]  = g[o] * rsqrtf(1.0f + 1e-5f);  // BN: g / sqrt(1+eps)
        bias[o] = b[o];
    }
}

// ---------------------------------------------------------------------------
// Farthest point sampling: one block (512 thr) per batch, point cloud in LDS.
// Emits centroid coordinates directly (new_xyz[b][i][0..2]).
// ---------------------------------------------------------------------------
__global__ __launch_bounds__(512)
void fps_kernel(const float* __restrict__ xyz, int stride, int N, int np,
                float* __restrict__ new_xyz) {
    __shared__ float sx[4096], sy[4096], sz[4096];
    __shared__ float redv[16];
    __shared__ int   redi[16];
    __shared__ int   sfar;
    int b = blockIdx.x, tid = threadIdx.x;

    for (int j = tid; j < N; j += 512) {
        const float* p = xyz + ((size_t)b * N + j) * stride;
        sx[j] = p[0]; sy[j] = p[1]; sz[j] = p[2];
    }
    if (tid == 0) sfar = 0;
    __syncthreads();

    const int ppt = N >> 9;                 // 4096->8, 512->1
    float px[8], py[8], pz[8], dd[8];
    for (int q = 0; q < ppt; ++q) {
        int j = tid + (q << 9);
        px[q] = sx[j]; py[q] = sy[j]; pz[q] = sz[j];
        dd[q] = 1e10f;
    }

    for (int it = 0; it < np; ++it) {
        int far = sfar;
        float cx = sx[far], cy = sy[far], cz = sz[far];
        if (tid == 0) {
            float* o = new_xyz + ((size_t)b * np + it) * 3;
            o[0] = cx; o[1] = cy; o[2] = cz;
        }
        float bv = -1.0f; int bi = 0x7fffffff;
        for (int q = 0; q < ppt; ++q) {
            int j = tid + (q << 9);
            float dx = px[q] - cx, dy = py[q] - cy, dz = pz[q] - cz;
            float d = dx * dx + dy * dy + dz * dz;
            dd[q] = fminf(dd[q], d);
            if (dd[q] > bv || (dd[q] == bv && j < bi)) { bv = dd[q]; bi = j; }
        }
        // wave32 argmax with first-index tie-break
        for (int off = 16; off >= 1; off >>= 1) {
            float ov = __shfl_xor(bv, off);
            int   oi = __shfl_xor(bi, off);
            if (ov > bv || (ov == bv && oi < bi)) { bv = ov; bi = oi; }
        }
        if ((tid & 31) == 0) { redv[tid >> 5] = bv; redi[tid >> 5] = bi; }
        __syncthreads();
        if (tid == 0) {
            float mv = redv[0]; int mi = redi[0];
            for (int w = 1; w < 16; ++w)
                if (redv[w] > mv || (redv[w] == mv && redi[w] < mi)) { mv = redv[w]; mi = redi[w]; }
            sfar = mi;
        }
        __syncthreads();
    }
}

// ---------------------------------------------------------------------------
// Ball query: one wave32 per centroid; ballot + prefix popcount collects the
// first `64` in-radius indices in ascending order, pads with the first index.
// ---------------------------------------------------------------------------
__global__ void ball_query_kernel(const float* __restrict__ xyz, int stride,
                                  const float* __restrict__ new_xyz,
                                  int N, int np, int nCentroids, float r2,
                                  int* __restrict__ out) {
    int gw = (blockIdx.x * blockDim.x + threadIdx.x) >> 5;
    int lane = threadIdx.x & 31;
    if (gw >= nCentroids) return;
    int b = gw / np;
    const float* nx = new_xyz + (size_t)gw * 3;
    float cx = nx[0], cy = nx[1], cz = nx[2];
    int* o = out + (size_t)gw * 64;

    int cnt = 0, first = 0;
    for (int base = 0; base < N && cnt < 64; base += 32) {
        int j = base + lane;
        const float* p = xyz + ((size_t)b * N + j) * stride;
        float dx = p[0] - cx, dy = p[1] - cy, dz = p[2] - cz;
        bool in = (dx * dx + dy * dy + dz * dz) < r2;
        unsigned mask = (unsigned)__ballot(in);
        int pos = cnt + __popc(mask & ((1u << lane) - 1u));
        if (in && pos < 64) o[pos] = j;
        if (cnt == 0 && mask) { int fl = __ffs((int)mask) - 1; first = __shfl(j, fl); }
        cnt += __popc(mask);
    }
    for (int p = cnt + lane; p < 64; p += 32) o[p] = first;
}

// ---------------------------------------------------------------------------
// Fused SA module: one workgroup (256 thr = 8 waves) per centroid.
// Gather 64 samples into LDS, run 3 WMMA layers (BN+ReLU) ping-ponging in LDS,
// column-max over the 64 samples, write f32 features.
// ---------------------------------------------------------------------------
__device__ __forceinline__ void mlp_layer_lds(const _Float16* in, _Float16* out,
                                              int K, int Cout,
                                              const _Float16* __restrict__ Wp,
                                              const float* __restrict__ scl,
                                              const float* __restrict__ bvec) {
    int wave = threadIdx.x >> 5;
    int lane = threadIdx.x & 31;
    int units = 4 * (Cout >> 4);            // 4 M-tiles x (Cout/16) N-tiles
    for (int u = wave; u < units; u += 8) {
        int m = u & 3, t = u >> 2;
        v8f acc = {0.f, 0.f, 0.f, 0.f, 0.f, 0.f, 0.f, 0.f};
        const _Float16* abase = in + (size_t)(m * 16) * K;
        const _Float16* wbase = Wp + (size_t)(t * 16) * K;
        for (int kb = 0; kb < K; kb += 32) {
            v16h af = load_frag(abase + kb, K);
            v16h bf = load_frag(wbase + kb, K);
            acc = wmma_f16(af, bf, acc);
        }
        int half = lane >> 4;
        int col = t * 16 + (lane & 15);
        float s = scl[col], bb = bvec[col];
        for (int i = 0; i < 8; ++i) {
            int row = m * 16 + half * 8 + i;
            float v = acc[i] * s + bb;
            v = v > 0.f ? v : 0.f;
            out[(size_t)row * Cout + col] = (_Float16)v;
        }
    }
}

__global__ __launch_bounds__(256)
void sa_fused_kernel(const float* __restrict__ xyz, int xyzStride,
                     const float* __restrict__ feat, int featStride, int Cf,
                     const int* __restrict__ idx,
                     const float* __restrict__ new_xyz,
                     const _Float16* W0, const float* s0, const float* b0,
                     const _Float16* W1, const float* s1, const float* b1,
                     const _Float16* W2, const float* s2, const float* b2,
                     int K0, int C1, int C2, int C3,
                     int np, int N,
                     float* __restrict__ outFeat) {
    __shared__ __align__(16) _Float16 bufA[64 * 160];   // X / layer-2 out
    __shared__ __align__(16) _Float16 bufB[64 * 256];   // layer-1 / layer-3 out
    __shared__ int sidx[64];

    int wg = blockIdx.x, b = wg / np, c = wg % np;
    int tid = threadIdx.x;
    if (tid < 64) sidx[tid] = idx[(size_t)wg * 64 + tid];
    const float* nx = new_xyz + (size_t)wg * 3;
    float cx = nx[0], cy = nx[1], cz = nx[2];
    (void)c;
    __syncthreads();

    // gather X tile (64 x K0), recentered xyz ++ features, zero pad
    for (int e = tid; e < 64 * K0; e += 256) {
        int srow = e / K0, k = e - srow * K0;
        int pi = sidx[srow];
        float v = 0.f;
        if (k < 3) {
            float cc = (k == 0) ? cx : ((k == 1) ? cy : cz);
            v = xyz[((size_t)b * N + pi) * xyzStride + k] - cc;
        } else if (k < 3 + Cf) {
            v = feat[((size_t)b * N + pi) * featStride + (k - 3)];
        }
        bufA[e] = (_Float16)v;
    }
    __syncthreads();

    mlp_layer_lds(bufA, bufB, K0, C1, W0, s0, b0);  __syncthreads();
    mlp_layer_lds(bufB, bufA, C1, C2, W1, s1, b1);  __syncthreads();
    mlp_layer_lds(bufA, bufB, C2, C3, W2, s2, b2);  __syncthreads();

    // max over the 64 samples, per channel
    if (tid < C3) {
        float m = 0.f;   // post-ReLU values are >= 0
        for (int r = 0; r < 64; ++r)
            m = fmaxf(m, (float)bufB[(size_t)r * C3 + tid]);
        outFeat[(size_t)wg * C3 + tid] = m;
    }
}

// ---------------------------------------------------------------------------
// Generic WMMA GEMM + BN + ReLU for SA3 layers: out = relu(X * Wp^T * s + b)
// workgroup = 64(M) x 128(N) tile, 8 waves, B-fragment reused across 4 M-tiles
// ---------------------------------------------------------------------------
__global__ __launch_bounds__(256)
void gemm_bn_relu_kernel(const _Float16* __restrict__ X,
                         const _Float16* __restrict__ Wp,
                         const float* __restrict__ scl,
                         const float* __restrict__ bvec,
                         _Float16* __restrict__ out,
                         int M, int Nn, int K) {
    int mtiles = M >> 6;
    int mg = blockIdx.x % mtiles;
    int ng = blockIdx.x / mtiles;
    int wave = threadIdx.x >> 5, lane = threadIdx.x & 31;
    int n0 = ng * 128 + wave * 16;

    v8f acc[4];
    for (int m = 0; m < 4; ++m) acc[m] = (v8f){0.f,0.f,0.f,0.f,0.f,0.f,0.f,0.f};

    const _Float16* abase = X  + (size_t)(mg * 64) * K;
    const _Float16* wbase = Wp + (size_t)n0 * K;
    for (int kb = 0; kb < K; kb += 32) {
        v16h bf = load_frag(wbase + kb, K);
        for (int m = 0; m < 4; ++m) {
            v16h af = load_frag(abase + (size_t)(m * 16) * K + kb, K);
            acc[m] = wmma_f16(af, bf, acc[m]);
        }
    }
    int half = lane >> 4;
    int col = n0 + (lane & 15);
    float s = scl[col], bb = bvec[col];
    for (int m = 0; m < 4; ++m)
        for (int i = 0; i < 8; ++i) {
            int row = mg * 64 + m * 16 + half * 8 + i;
            float v = acc[m][i] * s + bb;
            v = v > 0.f ? v : 0.f;
            out[(size_t)row * Nn + col] = (_Float16)v;
        }
}

// ---------------------------------------------------------------------------
// Small helper kernels
// ---------------------------------------------------------------------------
__global__ void build_x3_kernel(const float* __restrict__ nx2,
                                const float* __restrict__ f2,
                                _Float16* __restrict__ X3) {
    int t = blockIdx.x * blockDim.x + threadIdx.x;
    if (t >= 2048 * 288) return;
    int row = t / 288, k = t - row * 288;
    float v = 0.f;
    if (k < 3)        v = nx2[(size_t)row * 3 + k];
    else if (k < 259) v = f2[(size_t)row * 256 + (k - 3)];
    X3[t] = (_Float16)v;
}

__global__ void maxpool_kernel(const _Float16* __restrict__ H, float* __restrict__ out) {
    int t = blockIdx.x * blockDim.x + threadIdx.x;
    if (t >= 16 * 1024) return;
    int b = t >> 10, c = t & 1023;
    float m = 0.f;   // post-ReLU
    for (int r = 0; r < 128; ++r)
        m = fmaxf(m, (float)H[((size_t)(b * 128 + r)) * 1024 + c]);
    out[t] = m;
}

__global__ void fc_kernel(const float* __restrict__ X, const float* __restrict__ W,
                          const float* __restrict__ bias, float* __restrict__ out,
                          int M, int Nn, int K) {
    int t = blockIdx.x * blockDim.x + threadIdx.x;
    if (t >= M * Nn) return;
    int r = t / Nn, o = t - r * Nn;
    const float* x = X + (size_t)r * K;
    const float* w = W + (size_t)o * K;
    float s = bias ? bias[o] : 0.f;
    for (int k = 0; k < K; ++k) s += x[k] * w[k];
    out[t] = s;
}

__global__ __launch_bounds__(256)
void ln_silu_kernel(const float* __restrict__ h, const float* __restrict__ g,
                    const float* __restrict__ bvec, float* __restrict__ out) {
    __shared__ float ssum[256], ssq[256];
    int r = blockIdx.x, tid = threadIdx.x;
    float s = 0.f, q = 0.f;
    for (int c = tid; c < 1024; c += 256) {
        float v = h[(size_t)r * 1024 + c];
        s += v; q += v * v;
    }
    ssum[tid] = s; ssq[tid] = q; __syncthreads();
    for (int off = 128; off; off >>= 1) {
        if (tid < off) { ssum[tid] += ssum[tid + off]; ssq[tid] += ssq[tid + off]; }
        __syncthreads();
    }
    float mu = ssum[0] * (1.f / 1024.f);
    float var = ssq[0] * (1.f / 1024.f) - mu * mu;
    float inv = rsqrtf(var + 1e-5f);
    for (int c = tid; c < 1024; c += 256) {
        float v = (h[(size_t)r * 1024 + c] - mu) * inv * g[c] + bvec[c];
        out[(size_t)r * 1024 + c] = v / (1.f + expf(-v));
    }
}

// ---------------------------------------------------------------------------
// Host launch
// ---------------------------------------------------------------------------
extern "C" void kernel_launch(void* const* d_in, const int* in_sizes, int n_in,
                              void* d_out, int out_size, void* d_ws, size_t ws_size,
                              hipStream_t stream) {
    (void)in_sizes; (void)n_in; (void)out_size; (void)ws_size;
    const float* pc = (const float*)d_in[0];
    const float* sa_w[9] = {(const float*)d_in[1],(const float*)d_in[2],(const float*)d_in[3],
                            (const float*)d_in[10],(const float*)d_in[11],(const float*)d_in[12],
                            (const float*)d_in[19],(const float*)d_in[20],(const float*)d_in[21]};
    const float* sa_g[9] = {(const float*)d_in[4],(const float*)d_in[5],(const float*)d_in[6],
                            (const float*)d_in[13],(const float*)d_in[14],(const float*)d_in[15],
                            (const float*)d_in[22],(const float*)d_in[23],(const float*)d_in[24]};
    const float* sa_b[9] = {(const float*)d_in[7],(const float*)d_in[8],(const float*)d_in[9],
                            (const float*)d_in[16],(const float*)d_in[17],(const float*)d_in[18],
                            (const float*)d_in[25],(const float*)d_in[26],(const float*)d_in[27]};
    const float* fc1_w = (const float*)d_in[28];
    const float* ln_g  = (const float*)d_in[29];
    const float* ln_b  = (const float*)d_in[30];
    const float* fc2_w = (const float*)d_in[31];
    const float* fc2_b = (const float*)d_in[32];

    const int co[9]   = {64, 64, 128, 128, 128, 256, 256, 512, 1024};
    const int ci[9]   = {6,  64, 64,  131, 128, 128, 259, 256, 512};
    const int kpad[9] = {32, 64, 64,  160, 128, 128, 288, 256, 512};

    // workspace carve-out (256B aligned)
    char* ws = (char*)d_ws;
    size_t off = 0;
    auto alloc = [&](size_t bytes) { void* p = ws + off; off = (off + bytes + 255) & ~(size_t)255; return p; };

    _Float16* Wp[9]; float* scl[9]; float* bvec[9];
    for (int l = 0; l < 9; ++l) {
        Wp[l]   = (_Float16*)alloc((size_t)co[l] * kpad[l] * 2);
        scl[l]  = (float*)alloc((size_t)co[l] * 4);
        bvec[l] = (float*)alloc((size_t)co[l] * 4);
    }
    float* nxyz1 = (float*)alloc((size_t)16 * 512 * 3 * 4);
    float* nxyz2 = (float*)alloc((size_t)16 * 128 * 3 * 4);
    int*   idx1  = (int*)  alloc((size_t)16 * 512 * 64 * 4);
    int*   idx2  = (int*)  alloc((size_t)16 * 128 * 64 * 4);
    float* feat1 = (float*)alloc((size_t)16 * 512 * 128 * 4);
    float* feat2 = (float*)alloc((size_t)16 * 128 * 256 * 4);
    _Float16* X3  = (_Float16*)alloc((size_t)2048 * 288 * 2);
    _Float16* H31 = (_Float16*)alloc((size_t)2048 * 256 * 2);
    _Float16* H32 = (_Float16*)alloc((size_t)2048 * 512 * 2);
    _Float16* H33 = (_Float16*)alloc((size_t)2048 * 1024 * 2);
    float* fglob = (float*)alloc((size_t)16 * 1024 * 4);
    float* hfc   = (float*)alloc((size_t)16 * 1024 * 4);
    float* hln   = (float*)alloc((size_t)16 * 1024 * 4);

    // 0) weight prep (f32 -> padded f16 + BN scale/bias)
    for (int l = 0; l < 9; ++l) {
        int n = co[l] * kpad[l];
        prep_layer_kernel<<<(n + 255) / 256, 256, 0, stream>>>(
            sa_w[l], sa_g[l], sa_b[l], Wp[l], scl[l], bvec[l], co[l], ci[l], kpad[l]);
    }

    // 1) SA1: FPS -> ball query -> fused MLP+max
    fps_kernel<<<16, 512, 0, stream>>>(pc, 6, 4096, 512, nxyz1);
    {
        int nc = 16 * 512;
        ball_query_kernel<<<(nc * 32 + 255) / 256, 256, 0, stream>>>(
            pc, 6, nxyz1, 4096, 512, nc, 0.04f, idx1);
        sa_fused_kernel<<<nc, 256, 0, stream>>>(
            pc, 6, pc + 3, 6, 3, idx1, nxyz1,
            Wp[0], scl[0], bvec[0], Wp[1], scl[1], bvec[1], Wp[2], scl[2], bvec[2],
            32, 64, 64, 128, 512, 4096, feat1);
    }

    // 2) SA2
    fps_kernel<<<16, 512, 0, stream>>>(nxyz1, 3, 512, 128, nxyz2);
    {
        int nc = 16 * 128;
        ball_query_kernel<<<(nc * 32 + 255) / 256, 256, 0, stream>>>(
            nxyz1, 3, nxyz2, 512, 128, nc, 0.16f, idx2);
        sa_fused_kernel<<<nc, 256, 0, stream>>>(
            nxyz1, 3, feat1, 128, 128, idx2, nxyz2,
            Wp[3], scl[3], bvec[3], Wp[4], scl[4], bvec[4], Wp[5], scl[5], bvec[5],
            160, 128, 128, 256, 128, 512, feat2);
    }

    // 3) SA3: global MLP (WMMA GEMMs) + max over 128 points
    build_x3_kernel<<<(2048 * 288 + 255) / 256, 256, 0, stream>>>(nxyz2, feat2, X3);
    gemm_bn_relu_kernel<<<(2048 / 64) * (256 / 128), 256, 0, stream>>>(
        X3, Wp[6], scl[6], bvec[6], H31, 2048, 256, 288);
    gemm_bn_relu_kernel<<<(2048 / 64) * (512 / 128), 256, 0, stream>>>(
        H31, Wp[7], scl[7], bvec[7], H32, 2048, 512, 256);
    gemm_bn_relu_kernel<<<(2048 / 64) * (1024 / 128), 256, 0, stream>>>(
        H32, Wp[8], scl[8], bvec[8], H33, 2048, 1024, 512);
    maxpool_kernel<<<(16 * 1024 + 255) / 256, 256, 0, stream>>>(H33, fglob);

    // 4) head: fc1 -> LayerNorm -> SiLU -> fc2 (+bias)
    fc_kernel<<<(16 * 1024 + 255) / 256, 256, 0, stream>>>(
        fglob, fc1_w, nullptr, hfc, 16, 1024, 1024);
    ln_silu_kernel<<<16, 256, 0, stream>>>(hfc, ln_g, ln_b, hln);
    fc_kernel<<<(16 * 768 + 255) / 256, 256, 0, stream>>>(
        hln, fc2_w, fc2_b, (float*)d_out, 16, 768, 1024);
}
